// SelfAttention_13898514170451
// MI455X (gfx1250) — matrix operand order repaired and checked by reading, not compiled, not run on hardware
//
#include <hip/hip_runtime.h>

#define B_ 8
#define T_ 2048
#define C_ 1024
#define H_ 64

typedef __attribute__((ext_vector_type(16))) __bf16 v16bf;
typedef __attribute__((ext_vector_type(16))) float  v16f;
typedef __attribute__((ext_vector_type(8)))  __bf16 v8bf;
typedef __attribute__((ext_vector_type(8)))  float  v8f;

union BF16Frag {
  v16bf v;
  unsigned short u[16];
  uint4 q[2];
};

union F32x16 {
  v16f   v;
  float  f[16];
  float4 q4[4];
};

union BF16x8 {
  v8bf  v;
  uint4 q;
};

union F32x8 {
  v8f    v;
  float4 q4[2];
};

// Native f32 -> bf16 (RNE) via hardware cvt; returns raw bits for storage.
static __device__ __forceinline__ unsigned short bf16bits(float f) {
  __bf16 h = (__bf16)f;
  return __builtin_bit_cast(unsigned short, h);
}

static __device__ __forceinline__ v8f vzero() {
  v8f z;
  #pragma unroll
  for (int i = 0; i < 8; ++i) z[i] = 0.0f;
  return z;
}

static __device__ __forceinline__ v8f wmma_bf16(const BF16Frag& a, const BF16Frag& b, v8f c) {
  return __builtin_amdgcn_wmma_f32_16x16x32_bf16(false, a.v, false, b.v,
                                                 (short)0, c, false, false);
}

// A-layout fragment from f32: two 8-f32 runs at p and p+16, cvt to bf16.
static __device__ __forceinline__ void load_cvt_a(BF16Frag& dst, const float* p) {
  const float4* q1 = (const float4*)p;
  const float4* q2 = (const float4*)(p + 16);
  F32x16 t;
  t.q4[0] = q1[0]; t.q4[1] = q1[1];
  t.q4[2] = q2[0]; t.q4[3] = q2[1];
  dst.v = __builtin_convertvector(t.v, v16bf);   // v_cvt_pk_bf16_f32
}

// Two 8-bf16 runs at p and p+16 -> fragment (pure loads, no VALU).
static __device__ __forceinline__ void load_bf16x16(BF16Frag& dst, const unsigned short* p) {
  dst.q[0] = *(const uint4*)p;
  dst.q[1] = *(const uint4*)(p + 16);
}

// ---------------------------------------------------------------------------
// Kernel 0: one-shot weight conversion f32 -> bf16 into workspace.
// Wq is pre-scaled by H^-0.5 so the attention scale is fully folded.
// 3*H*C = 196608 elements = 96 blocks * 256 threads * 8 elems.
// ---------------------------------------------------------------------------
__global__ __launch_bounds__(256) void wcvt_kernel(
    const float* __restrict__ Wq, const float* __restrict__ Wk,
    const float* __restrict__ Wv, unsigned short* __restrict__ Wbf) {
  const int per = H_ * C_;                       // 65536, divisible by 8
  const int idx = (blockIdx.x * 256 + threadIdx.x) * 8;
  const int mat = idx / per;
  const int off = idx - mat * per;
  const float* src = (mat == 0) ? Wq : (mat == 1) ? Wk : Wv;
  const float scale = (mat == 0) ? 0.125f : 1.0f;

  F32x8 t;
  t.q4[0] = *(const float4*)(src + off);
  t.q4[1] = *(const float4*)(src + off + 4);
  #pragma unroll
  for (int i = 0; i < 8; ++i) t.v[i] *= scale;
  BF16x8 o;
  o.v = __builtin_convertvector(t.v, v8bf);
  *(uint4*)(Wbf + idx) = o.q;
}

// ---------------------------------------------------------------------------
// Kernel 1: projections qh = q@(0.125*Wq)^T, kh = k@Wk^T, vhT = (v@Wv^T)^T
// (bf16 out). One wave (32 threads) per (matrix, batch, 16-row t-tile).
// B fragments come pre-converted from Wbf: no B-side cvt VALU work, and the
// B registers are VMEM-written (loadcnt interlock, no WMMA->VALU WAR nops).
// ---------------------------------------------------------------------------
__global__ __launch_bounds__(32) void proj_kernel(
    const float* __restrict__ xq, const float* __restrict__ xk, const float* __restrict__ xv,
    const unsigned short* __restrict__ Wbf,
    unsigned short* __restrict__ qh, unsigned short* __restrict__ kh,
    unsigned short* __restrict__ vhT) {
  const int lane = threadIdx.x & 31;
  const int nCol = lane & 15;        // A: row index; B: column index
  const int half = lane >> 4;
  int tile = blockIdx.x;             // 0 .. 3*B*(T/16)-1
  const int mat = tile % 3;  tile /= 3;
  const int b  = tile / (T_ / 16);
  const int t0 = (tile % (T_ / 16)) * 16;

  const float* x = (mat == 0) ? xq : (mat == 1) ? xk : xv;

  v8f acc[4];
  #pragma unroll
  for (int j = 0; j < 4; ++j) acc[j] = vzero();

  const float* xrow = x + ((size_t)b * T_ + (t0 + nCol)) * C_;
  const int abase = half * 8;        // A K-offset per lane half
  const int kb16  = half * 8;        // B K-offset per lane half
  const unsigned short* wbase =
      Wbf + (size_t)mat * (H_ * C_) + (size_t)nCol * C_ + kb16;

  for (int c0 = 0; c0 < C_; c0 += 32) {
    // Prefetch the next K-step of the streaming activation row.
    __builtin_prefetch(xrow + c0 + 32 + abase, 0, 1);

    // A fragment: 16x32 tile of the input, lane = row, K = c (f32 -> bf16)
    BF16Frag a;
    load_cvt_a(a, xrow + c0 + abase);

    // B fragments: pre-converted bf16 weights, pure b128 loads.
    BF16Frag bb[4];
    load_bf16x16(bb[0], wbase + c0);
    load_bf16x16(bb[1], wbase + (size_t)16 * C_ + c0);
    load_bf16x16(bb[2], wbase + (size_t)32 * C_ + c0);
    load_bf16x16(bb[3], wbase + (size_t)48 * C_ + c0);

    acc[0] = wmma_bf16(a, bb[0], acc[0]);
    acc[1] = wmma_bf16(a, bb[1], acc[1]);
    acc[2] = wmma_bf16(a, bb[2], acc[2]);
    acc[3] = wmma_bf16(a, bb[3], acc[3]);
  }

  // Store D (C-layout: VGPR r -> row r + half*8, lane -> col)
  #pragma unroll
  for (int j = 0; j < 4; ++j) {
    #pragma unroll
    for (int r = 0; r < 8; ++r) {
      const int t = t0 + r + half * 8;
      const int h = 16 * j + nCol;
      const unsigned short val = bf16bits(acc[j][r]);
      if (mat == 0)      qh[((size_t)b * T_ + t) * H_ + h] = val;
      else if (mat == 1) kh[((size_t)b * T_ + t) * H_ + h] = val;
      else               vhT[((size_t)b * H_ + h) * T_ + t] = val;  // transposed
    }
  }
}

// ---------------------------------------------------------------------------
// Kernel 2: flash attention. One wave per (batch, 16-row q-tile), 32 keys/step.
// ---------------------------------------------------------------------------
__global__ __launch_bounds__(32) void attn_kernel(
    const unsigned short* __restrict__ qh, const unsigned short* __restrict__ kh,
    const unsigned short* __restrict__ vhT, const int* __restrict__ mask,
    float* __restrict__ out) {
  __shared__ alignas(16) unsigned short Plds[16 * 32];

  const int lane = threadIdx.x & 31;
  const int nCol = lane & 15;
  const int half = lane >> 4;
  const int b  = blockIdx.x / (T_ / 16);
  const int t0 = (blockIdx.x % (T_ / 16)) * 16;

  const float NEG_INF = -__builtin_inff();

  // q A-fragments (loop invariant): K = h, two frags cover h 0..31 / 32..63
  BF16Frag qa0, qa1;
  {
    const unsigned short* qrow = qh + ((size_t)b * T_ + t0 + nCol) * H_;
    const int ab = half * 8;
    qa0.q[0] = *(const uint4*)(qrow + ab);
    qa0.q[1] = *(const uint4*)(qrow + ab + 16);
    qa1.q[0] = *(const uint4*)(qrow + 32 + ab);
    qa1.q[1] = *(const uint4*)(qrow + 32 + ab + 16);
  }

  v8f O[4];
  #pragma unroll
  for (int j = 0; j < 4; ++j) O[j] = vzero();
  float m_i[8], l_i[8];
  #pragma unroll
  for (int r = 0; r < 8; ++r) { m_i[r] = NEG_INF; l_i[r] = 0.0f; }

  const int kend = t0 + 16;          // exclusive causal bound for this q-tile
  const int kb16 = half * 8;

  for (int kt = 0; kt < kend; kt += 32) {
    // --- S = q @ k^T for 32 keys; all 4 K-fragments loaded up front so the
    //     WMMA group has no internal WAR hazards and loads clause together ---
    const unsigned short* krow0 = kh + ((size_t)b * T_ + kt + nCol) * H_;
    const unsigned short* krow1 = krow0 + (size_t)16 * H_;
    BF16Frag kb[4];
    load_bf16x16(kb[0], krow0 + kb16);        // keys kt..kt+15,   h 0..31
    load_bf16x16(kb[1], krow0 + 32 + kb16);   // keys kt..kt+15,   h 32..63
    load_bf16x16(kb[2], krow1 + kb16);        // keys kt+16..+31,  h 0..31
    load_bf16x16(kb[3], krow1 + 32 + kb16);   // keys kt+16..+31,  h 32..63

    v8f S0 = vzero(), S1 = vzero();
    S0 = wmma_bf16(qa0, kb[0], S0);
    S1 = wmma_bf16(qa0, kb[2], S1);
    S0 = wmma_bf16(qa1, kb[1], S0);
    S1 = wmma_bf16(qa1, kb[3], S1);

    // --- mask + online softmax (scale pre-folded into Wq) ---
    const int key0 = kt + nCol;
    const int key1 = kt + 16 + nCol;
    const int mk0 = mask[b * T_ + key0];
    const int mk1 = mask[b * T_ + key1];

    #pragma unroll
    for (int r = 0; r < 8; ++r) {
      const int qrow = t0 + r + half * 8;
      float s0 = S0[r];
      float s1 = S1[r];
      if (key0 > qrow || mk0 == 0) s0 = NEG_INF;
      if (key1 > qrow || mk1 == 0) s1 = NEG_INF;

      float mx = fmaxf(s0, s1);
      #pragma unroll
      for (int off = 1; off < 16; off <<= 1)
        mx = fmaxf(mx, __shfl_xor(mx, off, 32));   // stays within 16-lane half

      const float mnew  = fmaxf(m_i[r], mx);
      const float alpha = (mnew == NEG_INF) ? 1.0f : __expf(m_i[r] - mnew);
      const float e0 = (s0 == NEG_INF) ? 0.0f : __expf(s0 - mnew);
      const float e1 = (s1 == NEG_INF) ? 0.0f : __expf(s1 - mnew);

      float rs = e0 + e1;
      #pragma unroll
      for (int off = 1; off < 16; off <<= 1)
        rs += __shfl_xor(rs, off, 32);

      l_i[r] = l_i[r] * alpha + rs;
      m_i[r] = mnew;
      O[0][r] *= alpha; O[1][r] *= alpha; O[2][r] *= alpha; O[3][r] *= alpha;

      const int prow = r + half * 8;
      Plds[prow * 32 + nCol]      = bf16bits(e0);
      Plds[prow * 32 + 16 + nCol] = bf16bits(e1);
    }
    __syncthreads();   // single-wave WG: orders LDS writes before reads

    // --- P (C-layout) -> A-layout bf16 fragment via LDS ---
    BF16Frag pa;
    {
      const unsigned short* pr = &Plds[nCol * 32];
      pa.q[0] = *(const uint4*)(pr + half * 8);
      pa.q[1] = *(const uint4*)(pr + half * 8 + 16);
    }

    // --- O += P @ V; all 4 V-fragments up front (contiguous from vhT) ---
    const unsigned short* vbase =
        vhT + ((size_t)b * H_ + nCol) * T_ + kt + kb16;
    BF16Frag vb[4];
    load_bf16x16(vb[0], vbase);
    load_bf16x16(vb[1], vbase + (size_t)16 * T_);
    load_bf16x16(vb[2], vbase + (size_t)32 * T_);
    load_bf16x16(vb[3], vbase + (size_t)48 * T_);

    O[0] = wmma_bf16(pa, vb[0], O[0]);
    O[1] = wmma_bf16(pa, vb[1], O[1]);
    O[2] = wmma_bf16(pa, vb[2], O[2]);
    O[3] = wmma_bf16(pa, vb[3], O[3]);

    __syncthreads();   // protect Plds before next iteration overwrites it
  }

  // --- normalize and store fp32 output ---
  #pragma unroll
  for (int r = 0; r < 8; ++r) {
    const float inv = 1.0f / l_i[r];
    const int t = t0 + r + half * 8;
    float* orow = out + ((size_t)b * T_ + t) * H_;
    orow[nCol]      = O[0][r] * inv;
    orow[16 + nCol] = O[1][r] * inv;
    orow[32 + nCol] = O[2][r] * inv;
    orow[48 + nCol] = O[3][r] * inv;
  }
}

// ---------------------------------------------------------------------------
extern "C" void kernel_launch(void* const* d_in, const int* in_sizes, int n_in,
                              void* d_out, int out_size, void* d_ws, size_t ws_size,
                              hipStream_t stream) {
  (void)in_sizes; (void)n_in; (void)out_size; (void)ws_size;
  const float* k    = (const float*)d_in[0];
  const float* q    = (const float*)d_in[1];
  const float* v    = (const float*)d_in[2];
  const int*   mask = (const int*)d_in[3];
  const float* Wk   = (const float*)d_in[4];
  const float* Wq   = (const float*)d_in[5];
  const float* Wv   = (const float*)d_in[6];
  float* out = (float*)d_out;

  const size_t elems = (size_t)B_ * T_ * H_;            // 1M elements
  unsigned short* qh  = (unsigned short*)d_ws;          // 2 MB
  unsigned short* kh  = qh + elems;                     // 2 MB
  unsigned short* vhT = kh + elems;                     // 2 MB (transposed [B,H,T])
  unsigned short* Wbf = vhT + elems;                    // 384 KB (bf16 weights)

  wcvt_kernel<<<(3 * H_ * C_) / (256 * 8), 256, 0, stream>>>(Wq, Wk, Wv, Wbf);
  proj_kernel<<<3 * B_ * (T_ / 16), 32, 0, stream>>>(q, k, v, Wbf, qh, kh, vhT);
  attn_kernel<<<B_ * (T_ / 16), 32, 0, stream>>>(qh, kh, vhT, mask, out);
}
